// TransformerLayer_57672820850698
// MI455X (gfx1250) — matrix operand order repaired and checked by reading, not compiled, not run on hardware
//
#include <hip/hip_runtime.h>
#include <hip/hip_bf16.h>

// ---------------- problem constants ----------------
#define BB 4
#define SS 2048
#define DD 1024
#define HH 16
#define HD 64
#define FF 4096
#define MM (BB*SS)   // 8192 rows

typedef unsigned short u16;
typedef __attribute__((ext_vector_type(16))) __bf16        v16bf;
typedef __attribute__((ext_vector_type(16))) unsigned short v16u;
typedef __attribute__((ext_vector_type(8)))  unsigned short v8u;
typedef __attribute__((ext_vector_type(8)))  float          v8f;

__device__ __forceinline__ u16 f2bf(float f) {
    unsigned int u = __float_as_uint(f);
    u += 0x7FFFu + ((u >> 16) & 1u);          // round-to-nearest-even
    return (u16)(u >> 16);
}

// A-fragment (16x32 bf16, M-major rows): lane<16 -> cols {k0..k0+7, k0+16..k0+23},
// lane>=16 -> +8.  Two 16-byte LDS loads.
__device__ __forceinline__ v16u ld_frag_a(const u16* lds, int stride, int rowbase,
                                          int lane, int kofs) {
    int row = rowbase + (lane & 15);
    const u16* p = lds + row * stride + kofs + ((lane & 16) ? 8 : 0);
    v8u lo = *(const v8u*)p;
    v8u hi = *(const v8u*)(p + 16);
    return __builtin_shufflevector(lo, hi, 0,1,2,3,4,5,6,7,8,9,10,11,12,13,14,15);
}

// B-fragment (32x16 bf16 stored [N][K]): lane<16 -> cols {k0..k0+7, k0+16..k0+23}, lane>=16 -> +8.
__device__ __forceinline__ v16u ld_frag_b(const u16* lds, int stride, int rowbase,
                                          int lane, int kofs) {
    int row = rowbase + (lane & 15);
    const u16* p = lds + row * stride + kofs + ((lane & 16) ? 8 : 0);
    v8u lo = *(const v8u*)p;
    v8u hi = *(const v8u*)(p + 16);
    return __builtin_shufflevector(lo, hi, 0,1,2,3,4,5,6,7,8,9,10,11,12,13,14,15);
}

__device__ __forceinline__ v8f wmma_bf16(v16u a, v16u b, v8f c) {
    return __builtin_amdgcn_wmma_f32_16x16x32_bf16(
        false, __builtin_bit_cast(v16bf, a),
        false, __builtin_bit_cast(v16bf, b),
        (short)0, c, false, false);
}

// ---------------- f32 [K,N] -> bf16 transposed [N,K] ----------------
// One-time per launch; hoists the weight transpose out of the GEMM hot loop so
// GEMM B-staging is pure b128 load/store.  Block handles a 32x32 tile.
__global__ __launch_bounds__(256) void cvt_transpose_kernel(const float* __restrict__ W,
                                                            u16* __restrict__ Wt,
                                                            int K, int N) {
    __shared__ u16 tile[32 * 34];
    const int t = threadIdx.x;
    const int n0 = blockIdx.x * 32, k0 = blockIdx.y * 32;
    const int c = t & 31, r0 = t >> 5;          // 8 rows per pass, 4 passes
    #pragma unroll
    for (int p = 0; p < 4; ++p) {
        const int r = r0 + p * 8;
        tile[r * 34 + c] = f2bf(W[(size_t)(k0 + r) * N + n0 + c]);   // coalesced read
    }
    __syncthreads();
    #pragma unroll
    for (int p = 0; p < 4; ++p) {
        const int r = r0 + p * 8;
        Wt[(size_t)(n0 + r) * K + k0 + c] = tile[c * 34 + r];        // coalesced write
    }
}

// ---------------- layernorm: f32 row -> bf16 row ----------------
__global__ __launch_bounds__(256) void ln_kernel(const float* __restrict__ x,
                                                 const float* __restrict__ g,
                                                 const float* __restrict__ bta,
                                                 u16* __restrict__ out) {
    __shared__ float rs[256], rq[256];
    const int t = threadIdx.x;
    const size_t row = blockIdx.x;
    const float* xr = x + row * DD;
    float a0 = xr[t*4+0], a1 = xr[t*4+1], a2 = xr[t*4+2], a3 = xr[t*4+3];
    rs[t] = a0 + a1 + a2 + a3;
    rq[t] = a0*a0 + a1*a1 + a2*a2 + a3*a3;
    __syncthreads();
    for (int s = 128; s > 0; s >>= 1) {
        if (t < s) { rs[t] += rs[t+s]; rq[t] += rq[t+s]; }
        __syncthreads();
    }
    const float mu = rs[0] * (1.0f / DD);
    const float var = rq[0] * (1.0f / DD) - mu * mu;
    const float rstd = rsqrtf(var + 1e-5f);
    u16* orow = out + row * DD;
    int c = t * 4;
    orow[c+0] = f2bf((a0 - mu) * rstd * g[c+0] + bta[c+0]);
    orow[c+1] = f2bf((a1 - mu) * rstd * g[c+1] + bta[c+1]);
    orow[c+2] = f2bf((a2 - mu) * rstd * g[c+2] + bta[c+2]);
    orow[c+3] = f2bf((a3 - mu) * rstd * g[c+3] + bta[c+3]);
}

// ---------------- WMMA GEMM: C = A(bf16)[M,K] * Wt(bf16)[N,K]^T + epilogue ----------
// Block tile 128x128, BK=32.  8 waves: 4 along M (32 rows) x 2 along N (64 cols).
// Per wave per K-tile: 2 A frags + 4 B frags -> 8 WMMAs.
// MODE 0: out_bf = acc
// MODE 1: out_f  = acc + resid
// MODE 2: out_bf = relu(acc + bias)
// MODE 3: out_f  = resid + relu(acc + bias)
template<int MODE>
__global__ __launch_bounds__(256) void gemm_bf16_kernel(
        const u16* __restrict__ A, const u16* __restrict__ Wt,
        const float* __restrict__ bias, const float* __restrict__ resid,
        u16* __restrict__ outb, float* __restrict__ outf,
        int M, int N, int K) {
    __shared__ __align__(16) u16 sa[128 * 40];   // A tile [128 M][32 K], stride 40
    __shared__ __align__(16) u16 sb[128 * 40];   // W tile [128 N][32 K], stride 40
    const int t = threadIdx.x;
    const int lane = t & 31, w = t >> 5;
    const int m0 = blockIdx.y * 128, n0 = blockIdx.x * 128;
    const int wm = (w & 3) * 32, wn = (w >> 2) * 64;
    v8f acc[2][4] = {};

    const int srow = t >> 1, scol = (t & 1) * 16;   // staging: 2 thr/row, 16 elems each
    const int nk = K >> 5;

    const u16* ga = A  + (size_t)(m0 + srow) * K + scol;
    const u16* gb = Wt + (size_t)(n0 + srow) * K + scol;
    u16* la = sa + srow * 40 + scol;
    u16* lb = sb + srow * 40 + scol;

    for (int kt = 0; kt < nk; ++kt) {
        const int kb = kt * 32;
        v8u va0 = *(const v8u*)(ga + kb);
        v8u va1 = *(const v8u*)(ga + kb + 8);
        v8u vb0 = *(const v8u*)(gb + kb);
        v8u vb1 = *(const v8u*)(gb + kb + 8);
        *(v8u*)(la)     = va0;
        *(v8u*)(la + 8) = va1;
        *(v8u*)(lb)     = vb0;
        *(v8u*)(lb + 8) = vb1;
        if (kt + 1 < nk) {           // gfx1250: global_prefetch_b8 for next tiles
            __builtin_prefetch(ga + kb + 32, 0, 0);
            __builtin_prefetch(gb + kb + 32, 0, 0);
        }
        __syncthreads();

        v16u af0 = ld_frag_a(sa, 40, wm,      lane, 0);
        v16u af1 = ld_frag_a(sa, 40, wm + 16, lane, 0);
        #pragma unroll
        for (int j = 0; j < 4; ++j) {
            v16u bf = ld_frag_b(sb, 40, wn + j * 16, lane, 0);
            acc[0][j] = wmma_bf16(af0, bf, acc[0][j]);
            acc[1][j] = wmma_bf16(af1, bf, acc[1][j]);
        }
        __syncthreads();
    }

    const int hi = (lane & 16) ? 8 : 0;
    #pragma unroll
    for (int i = 0; i < 2; ++i)
        #pragma unroll
        for (int j = 0; j < 4; ++j) {
            const int gcol = n0 + wn + j * 16 + (lane & 15);
            #pragma unroll
            for (int r = 0; r < 8; ++r) {
                const int grow = m0 + wm + i * 16 + r + hi;
                const size_t g = (size_t)grow * N + gcol;
                float v = acc[i][j][r];
                if (MODE == 0) {
                    outb[g] = f2bf(v);
                } else if (MODE == 1) {
                    outf[g] = v + resid[g];
                } else if (MODE == 2) {
                    v += bias[gcol];
                    outb[g] = f2bf(v > 0.f ? v : 0.f);
                } else {
                    v += bias[gcol];
                    outf[g] = resid[g] + (v > 0.f ? v : 0.f);
                }
            }
        }
}

// ---------------- flash attention: one WG per (b, h, 64-query block) ----------------
__global__ __launch_bounds__(256) void attn_kernel(const u16* __restrict__ Q,
                                                   const u16* __restrict__ Km,
                                                   const u16* __restrict__ V,
                                                   u16* __restrict__ O) {
    __shared__ __align__(16) u16 sq[64 * 72];   // [q][hd]
    __shared__ __align__(16) u16 sk[64 * 72];   // [key][hd]  (== B frag layout for QK^T)
    __shared__ __align__(16) u16 sv[64 * 72];   // [hd][key]  (transposed V, B frag for PV)
    __shared__ __align__(16) u16 sp[64 * 72];   // probs bf16 [q][key]
    __shared__ float ssc[64 * 68];              // scores f32
    __shared__ float sm[64], sl[64], sscale[64];

    const int t = threadIdx.x, lane = t & 31, w = t >> 5;
    const int mt = (w & 3) * 16;      // wave's 16 query rows
    const int ntp = (w >> 2) * 32;    // wave's 32 output cols (2 tiles)
    const int b = blockIdx.z, h = blockIdx.y, q0 = blockIdx.x * 64;
    const size_t headoff = ((size_t)b * SS) * DD + (size_t)h * HD;

    const int srow = t >> 2, scol = (t & 3) * 16;
    {   // stage Q once
        const u16* gq = Q + headoff + (size_t)(q0 + srow) * DD + scol;
        *(v8u*)(sq + srow * 72 + scol)     = *(const v8u*)gq;
        *(v8u*)(sq + srow * 72 + scol + 8) = *(const v8u*)(gq + 8);
    }
    if (t < 64) { sm[t] = -1e30f; sl[t] = 0.f; }
    v8f accO[2] = {};
    __syncthreads();

    const int hi = (lane & 16) ? 8 : 0;
    for (int kb = 0; kb < SS / 64; ++kb) {
        const int k0g = kb * 64;
        {   // stage K tile (row-major) and V tile (transposed)
            const u16* gk = Km + headoff + (size_t)(k0g + srow) * DD + scol;
            *(v8u*)(sk + srow * 72 + scol)     = *(const v8u*)gk;
            *(v8u*)(sk + srow * 72 + scol + 8) = *(const v8u*)(gk + 8);
            const u16* gv = V + headoff + (size_t)(k0g + srow) * DD + scol;
            v8u v0 = *(const v8u*)gv;
            v8u v1 = *(const v8u*)(gv + 8);
            #pragma unroll
            for (int i = 0; i < 8; ++i) {
                sv[(scol + i) * 72 + srow]     = v0[i];
                sv[(scol + 8 + i) * 72 + srow] = v1[i];
            }
        }
        __syncthreads();

        // scores = Q @ K^T  (K-dim = HD = 64 -> 2 WMMAs per 16x16 tile)
        #pragma unroll
        for (int j = 0; j < 2; ++j) {
            const int nt = ntp + j * 16;
            v8f s = {};
            s = wmma_bf16(ld_frag_a(sq, 72, mt, lane, 0),  ld_frag_b(sk, 72, nt, lane, 0),  s);
            s = wmma_bf16(ld_frag_a(sq, 72, mt, lane, 32), ld_frag_b(sk, 72, nt, lane, 32), s);
            #pragma unroll
            for (int r = 0; r < 8; ++r)
                ssc[(mt + r + hi) * 68 + nt + (lane & 15)] = s[r] * 0.125f;  // 1/sqrt(64)
        }
        __syncthreads();

        // online softmax over this 64-key block (one thread per query row)
        if (t < 64) {
            const float* sr = ssc + t * 68;
            float bm = -1e30f;
            #pragma unroll 8
            for (int k = 0; k < 64; ++k) bm = fmaxf(bm, sr[k]);
            const float mo = sm[t];
            const float mn = fmaxf(mo, bm);
            const float c = __expf(mo - mn);
            float ls = 0.f;
            u16* pr = sp + t * 72;
            #pragma unroll 8
            for (int k = 0; k < 64; ++k) {
                const float p = __expf(sr[k] - mn);
                pr[k] = f2bf(p);
                ls += p;
            }
            sl[t] = sl[t] * c + ls;
            sm[t] = mn;
            sscale[t] = c;
        }
        __syncthreads();

        // rescale running O, then O += P @ V
        #pragma unroll
        for (int j = 0; j < 2; ++j)
            #pragma unroll
            for (int r = 0; r < 8; ++r)
                accO[j][r] *= sscale[mt + r + hi];
        #pragma unroll
        for (int j = 0; j < 2; ++j) {
            const int nt = ntp + j * 16;
            accO[j] = wmma_bf16(ld_frag_a(sp, 72, mt, lane, 0),  ld_frag_b(sv, 72, nt, lane, 0),  accO[j]);
            accO[j] = wmma_bf16(ld_frag_a(sp, 72, mt, lane, 32), ld_frag_b(sv, 72, nt, lane, 32), accO[j]);
        }
        __syncthreads();
    }

    // normalize and write O as bf16 in [B,S,H*HD]
    #pragma unroll
    for (int j = 0; j < 2; ++j) {
        const int col = ntp + j * 16 + (lane & 15);
        #pragma unroll
        for (int r = 0; r < 8; ++r) {
            const int row = mt + r + hi;
            const float v = accO[j][r] / sl[row];
            O[headoff + (size_t)(q0 + row) * DD + col] = f2bf(v);
        }
    }
}

// ---------------- host-side orchestration ----------------
extern "C" void kernel_launch(void* const* d_in, const int* in_sizes, int n_in,
                              void* d_out, int out_size, void* d_ws, size_t ws_size,
                              hipStream_t stream) {
    (void)in_sizes; (void)n_in; (void)out_size; (void)ws_size;
    const float* x     = (const float*)d_in[0];
    const float* ln1_g = (const float*)d_in[1];
    const float* ln1_b = (const float*)d_in[2];
    const float* ln2_g = (const float*)d_in[3];
    const float* ln2_b = (const float*)d_in[4];
    const float* Wq    = (const float*)d_in[5];
    const float* Wk    = (const float*)d_in[6];
    const float* Wv    = (const float*)d_in[7];
    const float* Wo    = (const float*)d_in[8];
    const float* W1    = (const float*)d_in[9];
    const float* b1    = (const float*)d_in[10];
    const float* W2    = (const float*)d_in[11];
    const float* b2    = (const float*)d_in[12];
    float* out = (float*)d_out;

    char* ws = (char*)d_ws;
    const size_t DDsz  = (size_t)DD * DD;       // 1M elems
    const size_t DFsz  = (size_t)DD * FF;       // 4M elems
    const size_t ROWsz = (size_t)MM * DD;       // 8M elems
    // bf16 transposed weights [N,K]
    u16* wq_t = (u16*)(ws);
    u16* wk_t = wq_t + DDsz;
    u16* wv_t = wk_t + DDsz;
    u16* wo_t = wv_t + DDsz;
    u16* w1_t = wo_t + DDsz;                    // [FF, D]
    u16* w2_t = w1_t + DFsz;                    // [D, FF]
    // activations
    u16* h_bf  = w2_t + DFsz;                   // [M,D]
    u16* q_bf  = h_bf + ROWsz;
    u16* k_bf  = q_bf + ROWsz;
    u16* v_bf  = k_bf + ROWsz;
    u16* ao_bf = v_bf + ROWsz;                  // attention output [M,D]
    float* x1  = (float*)(ao_bf + ROWsz);       // residual stream [M,D] f32
    // reuse: ff1 overlays h/q/k/v (64 MB), h2 overlays ao
    u16* ff1_bf = h_bf;                         // [M,FF]
    u16* h2_bf  = ao_bf;                        // [M,D]

    // 1) weights -> bf16, transposed to [N,K]
    cvt_transpose_kernel<<<dim3(DD/32, DD/32), 256, 0, stream>>>(Wq, wq_t, DD, DD);
    cvt_transpose_kernel<<<dim3(DD/32, DD/32), 256, 0, stream>>>(Wk, wk_t, DD, DD);
    cvt_transpose_kernel<<<dim3(DD/32, DD/32), 256, 0, stream>>>(Wv, wv_t, DD, DD);
    cvt_transpose_kernel<<<dim3(DD/32, DD/32), 256, 0, stream>>>(Wo, wo_t, DD, DD);
    cvt_transpose_kernel<<<dim3(FF/32, DD/32), 256, 0, stream>>>(W1, w1_t, DD, FF);  // [D,FF] -> [FF,D]
    cvt_transpose_kernel<<<dim3(DD/32, FF/32), 256, 0, stream>>>(W2, w2_t, FF, DD);  // [FF,D] -> [D,FF]

    // 2) LN1
    ln_kernel<<<dim3(MM), 256, 0, stream>>>(x, ln1_g, ln1_b, h_bf);

    // 3) Q,K,V projections
    gemm_bf16_kernel<0><<<dim3(DD/128, MM/128), 256, 0, stream>>>(h_bf, wq_t, nullptr, nullptr, q_bf, nullptr, MM, DD, DD);
    gemm_bf16_kernel<0><<<dim3(DD/128, MM/128), 256, 0, stream>>>(h_bf, wk_t, nullptr, nullptr, k_bf, nullptr, MM, DD, DD);
    gemm_bf16_kernel<0><<<dim3(DD/128, MM/128), 256, 0, stream>>>(h_bf, wv_t, nullptr, nullptr, v_bf, nullptr, MM, DD, DD);

    // 4) attention
    attn_kernel<<<dim3(SS/64, HH, BB), 256, 0, stream>>>(q_bf, k_bf, v_bf, ao_bf);

    // 5) output projection + residual -> x1 (f32)
    gemm_bf16_kernel<1><<<dim3(DD/128, MM/128), 256, 0, stream>>>(ao_bf, wo_t, nullptr, x, nullptr, x1, MM, DD, DD);

    // 6) LN2
    ln_kernel<<<dim3(MM), 256, 0, stream>>>(x1, ln2_g, ln2_b, h2_bf);

    // 7) FFN GEMM1: relu(h2 @ W1 + b1) -> bf16
    gemm_bf16_kernel<2><<<dim3(FF/128, MM/128), 256, 0, stream>>>(h2_bf, w1_t, b1, nullptr, ff1_bf, nullptr, MM, FF, DD);

    // 8) FFN GEMM2: x1 + relu(ff1 @ W2 + b2) -> final f32 output
    gemm_bf16_kernel<3><<<dim3(DD/128, MM/128), 256, 0, stream>>>(ff1_bf, w2_t, b2, x1, nullptr, out, MM, DD, FF);
}
